// MultiHeadedAttention_44925357916344
// MI455X (gfx1250) — matrix operand order repaired and checked by reading, not compiled, not run on hardware
//
#include <hip/hip_runtime.h>

// ---------------- types ----------------
typedef __bf16 bf16;
typedef __attribute__((ext_vector_type(16))) __bf16 bf16x16;
typedef __attribute__((ext_vector_type(8)))  __bf16 bf16x8;
typedef __attribute__((ext_vector_type(8)))  float  f32x8;

union BF16Frag { bf16x16 v; bf16x8 h[2]; };

#define DIM      768
#define HEADS    12
#define D_HEAD   64
#define BATCH    8
#define SEQ      1024
#define TOKENS   (BATCH * SEQ)          // 8192
#define THREE    (3 * DIM)              // 2304
#define ATTN_SCALE 0.03608439182435161f // 768^-0.5 (reference uses full dim)

// ---------------- WMMA helper ----------------
__device__ __forceinline__ f32x8 wmma_bf16(bf16x16 a, bf16x16 b, f32x8 c) {
    return __builtin_amdgcn_wmma_f32_16x16x32_bf16(
        /*neg_a=*/false, a, /*neg_b=*/false, b,
        /*c_mod=*/(short)0, c, /*reuse_a=*/false, /*reuse_b=*/false);
}

// A fragment 16x32 bf16: lane l holds row (l&15); half=(l>>4) selects
// K range [half*16, half*16+16) (contiguous, matches B fragments)
__device__ __forceinline__ bf16x16 load_a_frag(const bf16* __restrict__ rowptr,
                                               int k0, int half) {
    BF16Frag u;
    const bf16* p = rowptr + k0 + half * 16;
    u.h[0] = *(const bf16x8*)(p);
    u.h[1] = *(const bf16x8*)(p + 8);
    return u.v;
}

// ---------------- CDNA5 async global->LDS copy (ASYNCcnt path) --------------
// GVS mode: mem_addr = SGPR_U64_base + VGPR_I32_byte_offset ; dsaddr = VGPR lds
__device__ __forceinline__ void async_b128(unsigned int lds_off,
                                           unsigned int gbyte_off,
                                           unsigned long long gbase) {
    asm volatile("global_load_async_to_lds_b128 %0, %1, %2"
                 :: "v"(lds_off), "v"(gbyte_off), "s"(gbase)
                 : "memory");
}
__device__ __forceinline__ void wait_async0() {
    asm volatile("s_wait_asynccnt 0x0" ::: "memory");
}

// ---------------- conversion kernels ----------------
__global__ void cvt_f32_to_bf16(const float* __restrict__ in,
                                bf16* __restrict__ out, int n) {
    int i = blockIdx.x * blockDim.x + threadIdx.x;
    if (i < n) out[i] = (bf16)in[i];
}

// out[c * rows + r] = in[r * cols + c]   (transpose + convert)
__global__ void cvt_f32_to_bf16_T(const float* __restrict__ in,
                                  bf16* __restrict__ out, int rows, int cols) {
    int i = blockIdx.x * blockDim.x + threadIdx.x;
    if (i < rows * cols) {
        int r = i / cols, c = i % cols;
        out[(size_t)c * rows + r] = (bf16)in[i];
    }
}

// ---------------- double-buffered LDS-staged 64x64 GEMM block ---------------
// A: row-major [M,K] bf16.  Bt: row-major [N,K] bf16 (K-contiguous per col).
// Block = 128 threads (4 waves); wave w computes rows [w*16, w*16+16) x 64 cols.
// Each K-stage (KC=64 deep): async-stage A(64x64) and B(64x64) tiles into LDS,
// double buffered, while WMMAs consume the previous stage from LDS.
template <int KDIM>
__device__ __forceinline__ void gemm_block_64x64(const bf16* __restrict__ A,
                                                 const bf16* __restrict__ Bt,
                                                 int rowblk, int colblk,
                                                 bf16 (*__restrict__ Atile)[64 * 64],
                                                 bf16 (*__restrict__ Btile)[64 * 64],
                                                 f32x8 acc[4]) {
    const int tid  = threadIdx.x;
    const int lane = tid & 31, wave = tid >> 5;
    const int half = lane >> 4, l16 = lane & 15;
    const unsigned long long abase = (unsigned long long)(uintptr_t)A;
    const unsigned long long bbase = (unsigned long long)(uintptr_t)Bt;

    auto stage = [&](int buf, int k0) {
        unsigned int aL = (unsigned int)(uintptr_t)(&Atile[buf][0]);
        unsigned int bL = (unsigned int)(uintptr_t)(&Btile[buf][0]);
#pragma unroll
        for (int i = 0; i < 4; ++i) {
            int c  = tid + i * 128;        // 512 x 16B chunks per 8KB tile
            int rr = c >> 3, sub = c & 7;  // row/col in tile, 8-elem chunk
            unsigned int loff = (unsigned int)((rr * 64 + sub * 8) * 2);
            unsigned int ga = (unsigned int)(((size_t)(rowblk + rr) * KDIM + k0 + sub * 8) * 2);
            unsigned int gb = (unsigned int)(((size_t)(colblk + rr) * KDIM + k0 + sub * 8) * 2);
            async_b128(aL + loff, ga, abase);
            async_b128(bL + loff, gb, bbase);
        }
    };

    stage(0, 0);
    wait_async0();
    __syncthreads();

    int cur = 0;
    const int NS = KDIM / 64;
    for (int ks = 0; ks < NS; ++ks) {
        if (ks + 1 < NS) stage(cur ^ 1, (ks + 1) * 64);  // prefetch next chunk

        const bf16* At = &Atile[cur][0];
        const bf16* Bl = &Btile[cur][0];
#pragma unroll
        for (int s = 0; s < 64; s += 32) {
            const bf16* arow = At + (wave * 16 + l16) * 64;
            bf16x16 a = load_a_frag(arow, s, half);
#pragma unroll
            for (int t = 0; t < 4; ++t) {
                bf16x16 b = *(const bf16x16*)(Bl + (t * 16 + l16) * 64 + s + half * 16);
                acc[t] = wmma_bf16(a, b, acc[t]);
            }
        }
        wait_async0();      // next-stage tiles landed in LDS
        __syncthreads();    // visible to all waves; cur buffer free to refill
        cur ^= 1;
    }
}

// ---------------- GEMM 1: qkv = x @ w_in, scatter into Q / K / V^T ----------
__global__ __launch_bounds__(128)
void gemm_qkv_kernel(const bf16* __restrict__ xb,      // [8192,768]
                     const bf16* __restrict__ winT,    // [2304,768]
                     bf16* __restrict__ qbuf,          // [96,1024,64]
                     bf16* __restrict__ kbuf,          // [96,1024,64]
                     bf16* __restrict__ vTbuf) {       // [96,64,1024]
    __shared__ bf16 Atile[2][64 * 64];
    __shared__ bf16 Btile[2][64 * 64];

    const int lane = threadIdx.x & 31, wave = threadIdx.x >> 5;
    const int half = lane >> 4, l16 = lane & 15;
    const int row0 = blockIdx.y * 64 + wave * 16;
    const int col0 = blockIdx.x * 64;

    f32x8 acc[4] = {};
    gemm_block_64x64<DIM>(xb, winT, blockIdx.y * 64, col0, Atile, Btile, acc);

#pragma unroll
    for (int t = 0; t < 4; ++t) {
#pragma unroll
        for (int r = 0; r < 8; ++r) {
            int row = row0 + r + 8 * half;          // token 0..8191
            int col = col0 + t * 16 + l16;          // 0..2303
            int sel  = col / DIM;
            int rem  = col - sel * DIM;
            int head = rem >> 6;
            int d    = rem & 63;
            int bb   = row >> 10;
            int n    = row & 1023;
            int bh   = bb * HEADS + head;
            bf16 bv = (bf16)acc[t][r];
            if (sel == 0)      qbuf[((size_t)bh * SEQ + n) * D_HEAD + d] = bv;
            else if (sel == 1) kbuf[((size_t)bh * SEQ + n) * D_HEAD + d] = bv;
            else               vTbuf[((size_t)bh * D_HEAD + d) * SEQ + n] = bv;
        }
    }
}

// ---------------- flash attention: 1 block = (bh, 64 query rows) ------------
__global__ __launch_bounds__(128)
void attn_kernel(const bf16* __restrict__ q,     // [96,1024,64]
                 const bf16* __restrict__ k,     // [96,1024,64]
                 const bf16* __restrict__ vT,    // [96,64,1024]
                 bf16* __restrict__ attn_out) {  // [8192,768]
    __shared__ bf16 plds[4][16][40];             // per-wave 16x32 P tile (+pad)

    const int lane = threadIdx.x & 31, wave = threadIdx.x >> 5;
    const int half = lane >> 4, l16 = lane & 15;
    const int bh    = blockIdx.y;
    const int batch = bh / HEADS;
    const int head  = bh - batch * HEADS;
    const int q0    = blockIdx.x * 64 + wave * 16;   // local query row base

    const bf16* qb = q  + (size_t)bh * SEQ * D_HEAD;
    const bf16* kb = k  + (size_t)bh * SEQ * D_HEAD;
    const bf16* vb = vT + (size_t)bh * D_HEAD * SEQ;

    // Q fragments (K-dim = 64 -> two 16x32 A fragments), reused all iterations
    const bf16* qrow = qb + (size_t)(q0 + l16) * D_HEAD;
    bf16x16 qa0 = load_a_frag(qrow, 0,  half);
    bf16x16 qa1 = load_a_frag(qrow, 32, half);

    f32x8 o[4] = {};
    float mrow[8], lrow[8];
#pragma unroll
    for (int r = 0; r < 8; ++r) { mrow[r] = -1e30f; lrow[r] = 0.0f; }

    for (int j0 = 0; j0 < SEQ; j0 += 32) {
        // ---- scores S = Q K^T for keys [j0, j0+32) : two 16x16 tiles ----
        const bf16* k0p = kb + (size_t)(j0 + l16) * D_HEAD + half * 16;
        const bf16* k1p = kb + (size_t)(j0 + 16 + l16) * D_HEAD + half * 16;
        bf16x16 kb0a = *(const bf16x16*)(k0p);
        bf16x16 kb0b = *(const bf16x16*)(k0p + 32);
        bf16x16 kb1a = *(const bf16x16*)(k1p);
        bf16x16 kb1b = *(const bf16x16*)(k1p + 32);

        f32x8 s0 = {}, s1 = {};
        s0 = wmma_bf16(qa0, kb0a, s0);
        s0 = wmma_bf16(qa1, kb0b, s0);
        s1 = wmma_bf16(qa0, kb1a, s1);
        s1 = wmma_bf16(qa1, kb1b, s1);

        // ---- online softmax (rows split lanes 0-15 / 16-31 per C layout) ----
#pragma unroll
        for (int r = 0; r < 8; ++r) {
            float a = s0[r] * ATTN_SCALE;
            float b = s1[r] * ATTN_SCALE;
            float t = fmaxf(a, b);
            t = fmaxf(t, __shfl_xor(t, 1, 32));
            t = fmaxf(t, __shfl_xor(t, 2, 32));
            t = fmaxf(t, __shfl_xor(t, 4, 32));
            t = fmaxf(t, __shfl_xor(t, 8, 32));
            float newm = fmaxf(mrow[r], t);
            float corr = __expf(mrow[r] - newm);
            mrow[r] = newm;
            float p0 = __expf(a - newm);
            float p1 = __expf(b - newm);
            float ps = p0 + p1;
            ps += __shfl_xor(ps, 1, 32);
            ps += __shfl_xor(ps, 2, 32);
            ps += __shfl_xor(ps, 4, 32);
            ps += __shfl_xor(ps, 8, 32);
            lrow[r] = lrow[r] * corr + ps;
            o[0][r] *= corr; o[1][r] *= corr; o[2][r] *= corr; o[3][r] *= corr;
            // stage P (bf16) into this wave's LDS slice in [row][key] order
            plds[wave][r + 8 * half][l16]      = (bf16)p0;
            plds[wave][r + 8 * half][16 + l16] = (bf16)p1;
        }

        // ---- reload P as A fragment (16 rows x 32 keys) ----
        BF16Frag pu;
        const bf16* prow = &plds[wave][l16][0];
        pu.h[0] = *(const bf16x8*)(prow + half * 16);
        pu.h[1] = *(const bf16x8*)(prow + half * 16 + 8);
        bf16x16 pa = pu.v;

        // ---- O += P @ V (V^T storage -> contiguous B fragments) ----
#pragma unroll
        for (int t2 = 0; t2 < 4; ++t2) {
            bf16x16 vfrag = *(const bf16x16*)(
                vb + (size_t)(t2 * 16 + l16) * SEQ + j0 + half * 16);
            o[t2] = wmma_bf16(pa, vfrag, o[t2]);
        }
    }

    // ---- normalize and write attn_out[token, head*64 + d] as bf16 ----
#pragma unroll
    for (int r = 0; r < 8; ++r) {
        float inv = 1.0f / lrow[r];
        int token = batch * SEQ + q0 + r + 8 * half;
        size_t base = (size_t)token * DIM + head * D_HEAD;
#pragma unroll
        for (int t2 = 0; t2 < 4; ++t2) {
            attn_out[base + t2 * 16 + l16] = (bf16)(o[t2][r] * inv);
        }
    }
}

// ---------------- GEMM 2: out = attn_out @ w_out + b_out (fp32 out) ---------
__global__ __launch_bounds__(128)
void gemm_out_kernel(const bf16* __restrict__ attn,   // [8192,768]
                     const bf16* __restrict__ woutT,  // [768,768]
                     const float* __restrict__ bias,  // [768]
                     float* __restrict__ out) {       // [8192,768]
    __shared__ bf16 Atile[2][64 * 64];
    __shared__ bf16 Btile[2][64 * 64];

    const int lane = threadIdx.x & 31, wave = threadIdx.x >> 5;
    const int half = lane >> 4, l16 = lane & 15;
    const int row0 = blockIdx.y * 64 + wave * 16;
    const int col0 = blockIdx.x * 64;

    f32x8 acc[4] = {};
    gemm_block_64x64<DIM>(attn, woutT, blockIdx.y * 64, col0, Atile, Btile, acc);

#pragma unroll
    for (int t = 0; t < 4; ++t) {
        int col = col0 + t * 16 + l16;
        float bv = bias[col];
#pragma unroll
        for (int r = 0; r < 8; ++r) {
            int row = row0 + r + 8 * half;
            out[(size_t)row * DIM + col] = acc[t][r] + bv;
        }
    }
}

// ---------------- launcher ----------------
extern "C" void kernel_launch(void* const* d_in, const int* in_sizes, int n_in,
                              void* d_out, int out_size, void* d_ws, size_t ws_size,
                              hipStream_t stream) {
    (void)in_sizes; (void)n_in; (void)out_size; (void)ws_size;
    const float* x     = (const float*)d_in[0];  // [8,1024,768]
    const float* w_in  = (const float*)d_in[1];  // [768,2304]
    const float* w_out = (const float*)d_in[2];  // [768,768]
    const float* b_out = (const float*)d_in[3];  // [768]
    float* out = (float*)d_out;

    // workspace layout (bf16 buffers)
    char* ws = (char*)d_ws;
    const size_t SZ_XB    = (size_t)TOKENS * DIM * sizeof(bf16);       // 12.6 MB
    const size_t SZ_WINT  = (size_t)THREE  * DIM * sizeof(bf16);       //  3.5 MB
    const size_t SZ_WOUTT = (size_t)DIM    * DIM * sizeof(bf16);       //  1.2 MB
    const size_t SZ_HEADB = (size_t)BATCH * HEADS * SEQ * D_HEAD * sizeof(bf16); // 12.6 MB
    size_t off = 0;
    bf16* xb    = (bf16*)(ws + off); off += SZ_XB;
    bf16* winT  = (bf16*)(ws + off); off += SZ_WINT;
    bf16* woutT = (bf16*)(ws + off); off += SZ_WOUTT;
    bf16* qbuf  = (bf16*)(ws + off); off += SZ_HEADB;
    bf16* kbuf  = (bf16*)(ws + off); off += SZ_HEADB;
    bf16* vTbuf = (bf16*)(ws + off); off += SZ_HEADB;
    bf16* attnb = (bf16*)(ws + off); off += SZ_XB;

    // 1) precision conversion (+ transposes for K-contiguous B fragments)
    {
        int n = TOKENS * DIM;
        cvt_f32_to_bf16<<<(n + 255) / 256, 256, 0, stream>>>(x, xb, n);
        int nw = DIM * THREE;
        cvt_f32_to_bf16_T<<<(nw + 255) / 256, 256, 0, stream>>>(w_in, winT, DIM, THREE);
        int no = DIM * DIM;
        cvt_f32_to_bf16_T<<<(no + 255) / 256, 256, 0, stream>>>(w_out, woutT, DIM, DIM);
    }

    // 2) QKV projection GEMM with head-layout scatter
    {
        dim3 grid(THREE / 64, TOKENS / 64);   // (36, 128)
        gemm_qkv_kernel<<<grid, 128, 0, stream>>>(xb, winT, qbuf, kbuf, vTbuf);
    }

    // 3) flash attention per (bh, 64-query tile)
    {
        dim3 grid(SEQ / 64, BATCH * HEADS);   // (16, 96)
        attn_kernel<<<grid, 128, 0, stream>>>(qbuf, kbuf, vTbuf, attnb);
    }

    // 4) output projection + bias
    {
        dim3 grid(DIM / 64, TOKENS / 64);     // (12, 128)
        gemm_out_kernel<<<grid, 128, 0, stream>>>(attnb, woutT, b_out, out);
    }
}
